// SelfAttention_1846835937921
// MI455X (gfx1250) — compile-verified, matrix operand order checked by reading, not run on hardware
//
#include <hip/hip_runtime.h>
#include <hip/hip_bf16.h>

#define EMB 256
#define KEY 64
#define BB  8
#define SS  4096

typedef __attribute__((ext_vector_type(16))) __bf16 v16bf;
typedef __attribute__((ext_vector_type(8)))  __bf16 v8bf;
typedef __attribute__((ext_vector_type(8)))  float  v8f;
typedef __attribute__((ext_vector_type(4)))  float  v4f;
typedef __attribute__((ext_vector_type(4)))  int    v4i;

union V16U { v16bf v; v8bf h[2]; };

// Detect the gfx1250 async global->LDS builtin (ASYNCcnt path).
#if defined(__has_builtin)
#  if __has_builtin(__builtin_amdgcn_global_load_async_to_lds_b128)
#    define HAVE_ASYNC_LDS 1
#  endif
#endif
#ifndef HAVE_ASYNC_LDS
#  define HAVE_ASYNC_LDS 0
#endif

__device__ __forceinline__ v8f wmma_bf16(v16bf a, v16bf b, v8f c) {
    // 8 args: (neg_a, A, neg_b, B, c_mod, C, reuse_a, reuse_b)
    return __builtin_amdgcn_wmma_f32_16x16x32_bf16(
        false, a, false, b, (short)0, c, false, false);
}

// Async 16-byte global->LDS copy (tracked by ASYNCcnt), with sync fallback.
__device__ __forceinline__ void copy16_g2l(const __bf16* gsrc, __bf16* ldst) {
#if HAVE_ASYNC_LDS
    __builtin_amdgcn_global_load_async_to_lds_b128(
        (__attribute__((address_space(1))) v4i*)gsrc,
        (__attribute__((address_space(3))) v4i*)ldst, 0, 0);
#else
    *(v8bf*)ldst = *(const v8bf*)gsrc;
#endif
}

__device__ __forceinline__ void wait_async0() {
#if HAVE_ASYNC_LDS
#  if defined(__has_builtin) && __has_builtin(__builtin_amdgcn_s_wait_asynccnt)
    __builtin_amdgcn_s_wait_asynccnt(0);
#  else
    asm volatile("s_wait_asynccnt 0x0" ::: "memory");
#  endif
#endif
}

// Load a 16x32 bf16 fragment from a row-major f32 matrix.
// A-layout: lane m=l&15 holds row (row0+m); elems 0..7 = K kbase..kbase+7,
// elems 8..15 = K 16+kbase..; kbase = (l>>4)*8.  The same addressing serves as
// the B operand of scores = Q*K^T because K is stored row-major [S][64].
__device__ __forceinline__ v16bf load_frag_rm(const float* p, int ld, int row0, int kb) {
    int l = threadIdx.x & 31;
    int r = row0 + (l & 15);
    int kbase = (l >> 4) << 3;
    const float* s = p + (size_t)r * ld + kb + kbase;
    v4f f0 = *(const v4f*)(s);
    v4f f1 = *(const v4f*)(s + 4);
    v4f f2 = *(const v4f*)(s + 16);
    v4f f3 = *(const v4f*)(s + 20);
    v16bf o;
    o[0]  = (__bf16)f0[0]; o[1]  = (__bf16)f0[1]; o[2]  = (__bf16)f0[2]; o[3]  = (__bf16)f0[3];
    o[4]  = (__bf16)f1[0]; o[5]  = (__bf16)f1[1]; o[6]  = (__bf16)f1[2]; o[7]  = (__bf16)f1[3];
    o[8]  = (__bf16)f2[0]; o[9]  = (__bf16)f2[1]; o[10] = (__bf16)f2[2]; o[11] = (__bf16)f2[3];
    o[12] = (__bf16)f3[0]; o[13] = (__bf16)f3[1]; o[14] = (__bf16)f3[2]; o[15] = (__bf16)f3[3];
    return o;
}

__device__ __forceinline__ v8f v8f_zero() {
    v8f z = {0.f, 0.f, 0.f, 0.f, 0.f, 0.f, 0.f, 0.f};
    return z;
}

// ---------------------------------------------------------------------------
// Kernel 0: pre-swizzle Wq/Wk/Wv ([256][64] f32) into bf16 B-fragment layout:
// [w(3)][kb(8)][nt(4)][lane(32)][e(16)]  (96 KB total; stays L2-resident)
// ---------------------------------------------------------------------------
__global__ void __launch_bounds__(256)
swizzle_w_kernel(const float* __restrict__ Wq, const float* __restrict__ Wk,
                 const float* __restrict__ Wv, __bf16* __restrict__ Wswz) {
    int gid   = blockIdx.x * 256 + threadIdx.x;   // 0 .. 3*16384-1
    int w     = gid >> 14;
    int rem   = gid & 16383;
    int kbIdx = rem >> 11;
    int nt    = (rem >> 9) & 3;
    int lane  = (rem >> 4) & 31;
    int e     = rem & 15;
    int kbase = (lane >> 4) << 3;
    int n     = nt * 16 + (lane & 15);
    int kk    = (e < 8) ? (kbase + e) : (16 + kbase + (e - 8));
    int k     = kbIdx * 32 + kk;
    const float* W = (w == 0) ? Wq : (w == 1) ? Wk : Wv;
    Wswz[gid] = (__bf16)W[k * KEY + n];
}

// ---------------------------------------------------------------------------
// Kernel 1: QKV projection.  Each wave owns 16 rows of x ([B*S,256]) and
// produces 16x64 of Q, K, V via 8 k-steps x 12 WMMAs (shared A fragment).
// Q is stored pre-scaled by 1/sqrt(KEY) with bias folded in.
// ---------------------------------------------------------------------------
__global__ void __launch_bounds__(256)
qkv_kernel(const float* __restrict__ x, const __bf16* __restrict__ Wswz,
           const float* __restrict__ bq, const float* __restrict__ bk,
           const float* __restrict__ bv,
           float* __restrict__ Q, float* __restrict__ K, float* __restrict__ V) {
    int wave = threadIdx.x >> 5;
    int l    = threadIdx.x & 31;
    int row0 = blockIdx.x * 128 + wave * 16;

    v8f accQ[4], accK[4], accV[4];
#pragma unroll
    for (int nt = 0; nt < 4; ++nt) { accQ[nt] = v8f_zero(); accK[nt] = v8f_zero(); accV[nt] = v8f_zero(); }

    for (int kb2 = 0; kb2 < 8; ++kb2) {
        v16bf a = load_frag_rm(x, EMB, row0, kb2 * 32);
#pragma unroll
        for (int nt = 0; nt < 4; ++nt) {
            v16bf fq = *(const v16bf*)(Wswz + ((((0 * 8 + kb2) * 4 + nt) * 32 + l) * 16));
            v16bf fk = *(const v16bf*)(Wswz + ((((1 * 8 + kb2) * 4 + nt) * 32 + l) * 16));
            v16bf fv = *(const v16bf*)(Wswz + ((((2 * 8 + kb2) * 4 + nt) * 32 + l) * 16));
            accQ[nt] = wmma_bf16(a, fq, accQ[nt]);
            accK[nt] = wmma_bf16(a, fk, accK[nt]);
            accV[nt] = wmma_bf16(a, fv, accV[nt]);
        }
    }

    int rowsub = (l >> 4) * 8;
#pragma unroll
    for (int nt = 0; nt < 4; ++nt) {
        int n = nt * 16 + (l & 15);
        float vbq = bq[n], vbk = bk[n], vbv = bv[n];
#pragma unroll
        for (int r = 0; r < 8; ++r) {
            size_t idx = (size_t)(row0 + r + rowsub) * KEY + n;
            Q[idx] = (accQ[nt][r] + vbq) * 0.125f;   // fold 1/sqrt(64)
            K[idx] = accK[nt][r] + vbk;
            V[idx] = accV[nt][r] + vbv;
        }
    }
}

// ---------------------------------------------------------------------------
// Kernel 1b: swizzle V ([B*S,64] f32) into bf16 B-fragment tiles:
// [b(8)][kt(128)][nt(4)][lane(32)][e(16)]  (4 MB).  Each 32x64 V tile becomes
// a verbatim 4 KB block that kernel 3 can async-DMA into LDS untouched.
// ---------------------------------------------------------------------------
__global__ void __launch_bounds__(256)
swizzle_v_kernel(const float* __restrict__ V, __bf16* __restrict__ Vswz) {
    int gid  = blockIdx.x * 256 + threadIdx.x;   // 0 .. 8*128*2048-1
    int b    = gid >> 18;
    int rem  = gid & 262143;
    int kt   = rem >> 11;
    int rem2 = rem & 2047;
    int nt   = rem2 >> 9;
    int lane = (rem2 >> 4) & 31;
    int e    = rem2 & 15;
    int kbase = (lane >> 4) << 3;
    int n    = nt * 16 + (lane & 15);
    int kk   = (e < 8) ? (kbase + e) : (16 + kbase + (e - 8));
    Vswz[gid] = (__bf16)V[((size_t)b * SS + kt * 32 + kk) * KEY + n];
}

// ---------------------------------------------------------------------------
// Kernel 2: per-(b, k-column) online softmax stats over the QUERY axis.
// Each wave owns 16 columns, streams all 256 q-tiles with 2 WMMAs/tile,
// keeps running (max, sum) in registers, writes m[] and 1/Z[].
// ---------------------------------------------------------------------------
__global__ void __launch_bounds__(256)
colstats_kernel(const float* __restrict__ Q, const float* __restrict__ K,
                float* __restrict__ marr, float* __restrict__ zinv) {
    int wave = threadIdx.x >> 5;
    int l    = threadIdx.x & 31;
    int b    = blockIdx.y;
    int c0   = blockIdx.x * 128 + wave * 16;
    const float* Qb = Q + (size_t)b * SS * KEY;
    const float* Kb = K + (size_t)b * SS * KEY;

    // K fragments are loop-invariant for this wave's columns.
    v16bf kf0 = load_frag_rm(Kb, KEY, c0, 0);
    v16bf kf1 = load_frag_rm(Kb, KEY, c0, 32);

    float runM = -__builtin_inff();
    float runZ = 0.f;

    for (int q0 = 0; q0 < SS; q0 += 16) {
        v16bf a0 = load_frag_rm(Qb, KEY, q0, 0);
        v16bf a1 = load_frag_rm(Qb, KEY, q0, 32);
        v8f s = v8f_zero();
        s = wmma_bf16(a0, kf0, s);
        s = wmma_bf16(a1, kf1, s);

        float mt = s[0];
#pragma unroll
        for (int r = 1; r < 8; ++r) mt = fmaxf(mt, s[r]);
        mt = fmaxf(mt, __shfl_xor(mt, 16, 32));      // combine the two row-halves
        float nm = fmaxf(runM, mt);
        float part = 0.f;
#pragma unroll
        for (int r = 0; r < 8; ++r) part += __expf(s[r] - nm);
        part += __shfl_xor(part, 16, 32);
        runZ = runZ * __expf(runM - nm) + part;
        runM = nm;
    }

    if (l < 16) {
        marr[(size_t)b * SS + c0 + l] = runM;
        zinv[(size_t)b * SS + c0 + l] = 1.0f / runZ;
    }
}

// ---------------------------------------------------------------------------
// Kernel 3: output.  Each wave owns 16 q-rows; loop over k in steps of 32:
//   1) kick an async 4 KB DMA of the pre-swizzled V tile into LDS (ASYNCcnt)
//   2) recompute score tiles (WMMA), p = exp(s-m[k])*Zinv[k]
//   3) transpose P through per-wave LDS into A-fragment layout
//   4) s_wait_asynccnt + barrier, then WMMA-accumulate P @ V from LDS.
// The async copy overlaps with step 2/3's WMMAs and exp work.
// ---------------------------------------------------------------------------
__global__ void __launch_bounds__(256)
attn_out_kernel(const float* __restrict__ Q, const float* __restrict__ K,
                const __bf16* __restrict__ Vswz, const float* __restrict__ marr,
                const float* __restrict__ zinv, float* __restrict__ out) {
    __shared__ __align__(32) __bf16 ldsV[4 * 32 * 16];  // 4 KB: V 32x64 tile, frag layout
    __shared__ __align__(32) __bf16 ldsP[8][16 * 32];   // 8 KB: per-wave P transpose

    int wave = threadIdx.x >> 5;
    int l    = threadIdx.x & 31;
    int tid  = threadIdx.x;
    int b    = blockIdx.y;
    int q0   = blockIdx.x * 128 + wave * 16;
    const float* Qb = Q + (size_t)b * SS * KEY;
    const float* Kb = K + (size_t)b * SS * KEY;
    const __bf16* Vswb = Vswz + (size_t)b * 128 * 2048;

    v16bf aq0 = load_frag_rm(Qb, KEY, q0, 0);    // loop-invariant Q fragments
    v16bf aq1 = load_frag_rm(Qb, KEY, q0, 32);

    v8f accO[4];
#pragma unroll
    for (int nt = 0; nt < 4; ++nt) accO[nt] = v8f_zero();

    int m      = l & 15;
    int kbase  = (l >> 4) << 3;
    int rowsub = (l >> 4) * 8;

    for (int c = 0; c < SS; c += 32) {
        __syncthreads();   // all waves done reading ldsV of the previous tile

        // 1) Launch async copy of this 4 KB swizzled V tile (16 B per thread).
        copy16_g2l(Vswb + (size_t)(c >> 5) * 2048 + tid * 8, &ldsV[tid * 8]);

        // 2) Two 16-col score tiles -> normalized P written transposed to LDS.
#pragma unroll
        for (int t = 0; t < 2; ++t) {
            v16bf kf0 = load_frag_rm(Kb, KEY, c + t * 16, 0);
            v16bf kf1 = load_frag_rm(Kb, KEY, c + t * 16, 32);
            v8f s = v8f_zero();
            s = wmma_bf16(aq0, kf0, s);
            s = wmma_bf16(aq1, kf1, s);
            int col  = c + t * 16 + (l & 15);
            float mc = marr[(size_t)b * SS + col];
            float zc = zinv[(size_t)b * SS + col];
#pragma unroll
            for (int r = 0; r < 8; ++r) {
                float p = __expf(s[r] - mc) * zc;
                ldsP[wave][(r + rowsub) * 32 + t * 16 + (l & 15)] = (__bf16)p;
            }
        }

        // 3) Re-read P as an A fragment (16 q x 32 k) — same-wave LDS roundtrip.
        V16U pu;
        pu.h[0] = *(const v8bf*)&ldsP[wave][m * 32 + kbase];
        pu.h[1] = *(const v8bf*)&ldsP[wave][m * 32 + 16 + kbase];

        // 4) Wait for this wave's async tile, then barrier so all copies land.
        wait_async0();
        __syncthreads();

#pragma unroll
        for (int nt = 0; nt < 4; ++nt) {
            v16bf vf = *(const v16bf*)&ldsV[(nt * 32 + l) * 16];
            accO[nt] = wmma_bf16(pu.v, vf, accO[nt]);
        }
    }

#pragma unroll
    for (int nt = 0; nt < 4; ++nt) {
        int n = nt * 16 + (l & 15);
#pragma unroll
        for (int r = 0; r < 8; ++r)
            out[(size_t)(b * SS + q0 + r + rowsub) * KEY + n] = accO[nt][r];
    }
}

// ---------------------------------------------------------------------------
extern "C" void kernel_launch(void* const* d_in, const int* in_sizes, int n_in,
                              void* d_out, int out_size, void* d_ws, size_t ws_size,
                              hipStream_t stream) {
    const float* x  = (const float*)d_in[0];
    const float* Wq = (const float*)d_in[1];
    const float* bq = (const float*)d_in[2];
    const float* Wk = (const float*)d_in[3];
    const float* bk = (const float*)d_in[4];
    const float* Wv = (const float*)d_in[5];
    const float* bv = (const float*)d_in[6];
    float* out = (float*)d_out;

    // Workspace layout (~28.3 MB): Q | K | V | m | 1/Z | Wswz(bf16) | Vswz(bf16)
    float* Q    = (float*)d_ws;
    float* K    = Q + (size_t)BB * SS * KEY;
    float* V    = K + (size_t)BB * SS * KEY;
    float* marr = V + (size_t)BB * SS * KEY;
    float* zinv = marr + (size_t)BB * SS;
    __bf16* Wswz = (__bf16*)(zinv + (size_t)BB * SS);
    __bf16* Vswz = Wswz + (size_t)3 * 16384;

    swizzle_w_kernel<<<192, 256, 0, stream>>>(Wq, Wk, Wv, Wswz);
    qkv_kernel<<<(BB * SS) / 128, 256, 0, stream>>>(x, Wswz, bq, bk, bv, Q, K, V);
    swizzle_v_kernel<<<(BB * 128 * 2048) / 256, 256, 0, stream>>>(V, Vswz);
    colstats_kernel<<<dim3(SS / 128, BB), 256, 0, stream>>>(Q, K, marr, zinv);
    attn_out_kernel<<<dim3(SS / 128, BB), 256, 0, stream>>>(Q, K, Vswz, marr, zinv, out);
}
